// CustomGraphNet_21010980012131
// MI455X (gfx1250) — matrix operand order repaired and checked by reading, not compiled, not run on hardware
//
#include <hip/hip_runtime.h>

// ---------------------------------------------------------------------------
// MI455X (gfx1250) implementation of the graph-transformer middle block.
// All matrix contractions use v_wmma_f32_16x16x32_bf16 (fp32 accumulate).
// adj is binary -> exact in bf16. Weights/activations cast to bf16 once.
// GEMM uses a 2-stage LDS ping-pong: A tiles via global_load_async_to_lds_b128
// (ASYNCcnt), B tiles via register-staged transpose, prefetch 2 tiles ahead.
// ---------------------------------------------------------------------------

typedef __attribute__((ext_vector_type(16))) __bf16 v16bf;
typedef __attribute__((ext_vector_type(8)))  float  v8f;

union FragB  { unsigned int u[8]; v16bf v; };
union FragAcc{ float f[8]; v8f v; };
union U4     { uint4 q; unsigned short s[8]; };

enum { FLAG_BIAS = 1, FLAG_RESID = 2, FLAG_BVEC = 4,
       FLAG_WF32 = 8, FLAG_WBF = 16, FLAG_WBFT = 32 };

__device__ inline unsigned short f2bf(float f) {
  unsigned int u = __float_as_uint(f);
  unsigned int r = u + 0x7fffu + ((u >> 16) & 1u);   // round-to-nearest-even
  return (unsigned short)(r >> 16);
}

__device__ inline v8f wmma_bf16(v16bf a, v16bf b, v8f c) {
  return __builtin_amdgcn_wmma_f32_16x16x32_bf16(false, a, false, b, (short)0, c,
                                                 false, false);
}

// CDNA5 async DMA: 16B per lane, global -> LDS, tracked by ASYNCcnt.
__device__ inline void async_ld_b128(unsigned lds_off, const void* gptr) {
  asm volatile("global_load_async_to_lds_b128 %0, %1, off"
               :: "v"(lds_off), "v"(gptr) : "memory");
}
__device__ inline void wait_asynccnt0() {
  asm volatile("s_wait_asynccnt 0x0" ::: "memory");
}

// ---------------------------------------------------------------------------
// Generic bf16 GEMM: C[M,N] = A[M,K] @ B[K,N] (+bias +resid +bvec), batched.
// Tiles: 128x128x32, 256 threads = 8 waves (2x4 grid), each wave 64x32.
// 2-deep software pipeline:  async-DMA A(t+1) + reg-load B(t+1)  ||  WMMA(t).
// All problem dims are multiples of the tile sizes (no edge code).
// ---------------------------------------------------------------------------
template<int FLAGS>
__global__ __launch_bounds__(256) void gemm_bf16_kernel(
    const unsigned short* __restrict__ A, long sAb, int lda,
    const unsigned short* __restrict__ Bm, long sBb, int ldb,
    const float* __restrict__ bias,
    const float* __restrict__ resid, long sRb,
    const float* __restrict__ bvec,
    float* __restrict__ Cf, long sCf,
    unsigned short* __restrict__ Cb, long sCb,
    int Nn, int K)
{
  // A: 80B rows -> async b128 stores stay 16B-aligned; row stride 20 dwords
  //    (gcd(20,64)=4) spreads the 16 lanes of a half over 16 distinct banks.
  __shared__ unsigned short As[2][128][40];
  __shared__ unsigned short Bs[2][128][34];   // transposed: [n][k]

  const int tid  = threadIdx.x;
  const int wave = tid >> 5, lane = tid & 31;
  const int half = lane >> 4, l16 = lane & 15;
  const int wm = wave >> 2, wn = wave & 3;
  const int m0 = blockIdx.y * 128, n0 = blockIdx.x * 128;
  const long bz = blockIdx.z;
  const unsigned short* Ap = A + bz * sAb;
  const unsigned short* Bp = Bm + bz * sBb;

  const unsigned asbase = (unsigned)(unsigned long long)(void*)&As[0][0][0];

  FragAcc acc[4][2];
#pragma unroll
  for (int i = 0; i < 4; ++i)
#pragma unroll
    for (int j = 0; j < 2; ++j)
#pragma unroll
      for (int r = 0; r < 8; ++r) acc[i][j].f[r] = 0.f;

  // per-thread copy geometry (2 x 16B chunks each for A and B)
  const int a_row = tid >> 2,  a_c16 = (tid & 3) * 8;     // + it*64 rows
  const int b_kr  = tid >> 4,  b_n8  = (tid & 15) * 8;    // + it*16 k-rows

  auto issueA = [&](int kt, int buf) {
#pragma unroll
    for (int it = 0; it < 2; ++it) {
      int row = a_row + it * 64;
      unsigned lds = asbase + (unsigned)(buf * 10240 + row * 80 + a_c16 * 2);
      async_ld_b128(lds, Ap + (long)(m0 + row) * lda + kt + a_c16);
    }
  };
  auto loadBregs = [&](int kt, U4* d) {
#pragma unroll
    for (int it = 0; it < 2; ++it) {
      int kr = b_kr + it * 16;
      d[it].q = *reinterpret_cast<const uint4*>(Bp + (long)(kt + kr) * ldb + n0 + b_n8);
    }
  };
  auto storeB = [&](const U4* d, int buf) {
#pragma unroll
    for (int it = 0; it < 2; ++it) {
      int kr = b_kr + it * 16;
#pragma unroll
      for (int e = 0; e < 8; ++e) Bs[buf][b_n8 + e][kr] = d[it].s[e];
    }
  };

  const int T = K >> 5;
  U4 breg[2];

  // prologue: stage tile 0
  loadBregs(0, breg);
  issueA(0, 0);
  storeB(breg, 0);
  wait_asynccnt0();
  __syncthreads();

  for (int t = 0; t < T; ++t) {
    const int cur = t & 1, nxt = cur ^ 1;
    const bool more = (t + 1 < T);
    if (more) {
      issueA((t + 1) * 32, nxt);              // ASYNCcnt DMA overlaps WMMAs
      loadBregs((t + 1) * 32, breg);
      if (t + 2 < T)                          // -> global_prefetch_b8
        __builtin_prefetch(Bp + (long)((t + 2) * 32 + b_kr) * ldb + n0 + b_n8, 0, 0);
    }

    // fragments (contiguous k = 16*half + s, same order in A and B)
    FragB af[4], bfrag[2];
#pragma unroll
    for (int mt = 0; mt < 4; ++mt) {
      const unsigned short* ap = &As[cur][wm * 64 + mt * 16 + l16][0];
#pragma unroll
      for (int v2 = 0; v2 < 4; ++v2) {
        af[mt].u[v2]     = *(const unsigned int*)(ap + 2 * v2 + half * 16);
        af[mt].u[4 + v2] = *(const unsigned int*)(ap + 8 + 2 * v2 + half * 16);
      }
    }
#pragma unroll
    for (int nt = 0; nt < 2; ++nt) {
      const unsigned short* bp = &Bs[cur][wn * 32 + nt * 16 + l16][0];
#pragma unroll
      for (int v2 = 0; v2 < 8; ++v2)
        bfrag[nt].u[v2] = *(const unsigned int*)(bp + 2 * v2 + half * 16);
    }
#pragma unroll
    for (int mt = 0; mt < 4; ++mt)
#pragma unroll
      for (int nt = 0; nt < 2; ++nt)
        acc[mt][nt].v = wmma_bf16(af[mt].v, bfrag[nt].v, acc[mt][nt].v);

    if (more) {
      storeB(breg, nxt);
      wait_asynccnt0();                       // A(t+1) resident before barrier
    }
    __syncthreads();
  }

  // epilogue: C layout VGPR r <-> M = r (+8 upper half), N = lane&15
#pragma unroll
  for (int mt = 0; mt < 4; ++mt)
#pragma unroll
    for (int nt = 0; nt < 2; ++nt) {
      int nn = n0 + wn * 32 + nt * 16 + l16;
      float bi = 0.f;
      if constexpr (FLAGS & FLAG_BIAS) bi = bias[nn];
#pragma unroll
      for (int r = 0; r < 8; ++r) {
        int mm = m0 + wm * 64 + mt * 16 + r + half * 8;
        float val = acc[mt][nt].f[r] + bi;
        if constexpr (FLAGS & FLAG_BVEC)  val += bvec[(long)(mm >> 11) * Nn + nn];
        if constexpr (FLAGS & FLAG_RESID) val += resid[bz * sRb + (long)mm * Nn + nn];
        if constexpr (FLAGS & FLAG_WF32)  Cf[bz * sCf + (long)mm * Nn + nn] = val;
        if constexpr (FLAGS & FLAG_WBF)   Cb[bz * sCb + (long)mm * Nn + nn] = f2bf(val);
        if constexpr (FLAGS & FLAG_WBFT)  // vT[b, c, n] for attention B-frag loads
          Cb[(((long)(mm >> 11) << 8) + nn) * 2048 + (mm & 2047)] = f2bf(val);
      }
    }
}

// ---------------------------------------------------------------------------
// Flash-style masked attention. grid=(N/128, H=8, B=2), 256 thr = 8 waves,
// each wave owns 16 query rows. d_head = 32 -> one 16x16x32 WMMA per score
// tile, two j-tiles per step, p routed through LDS (C-layout -> A-layout).
// ---------------------------------------------------------------------------
__global__ __launch_bounds__(256) void attn_kernel(
    const unsigned short* __restrict__ q,
    const unsigned short* __restrict__ k,
    const unsigned short* __restrict__ vT,
    const float* __restrict__ adj,
    unsigned short* __restrict__ o)
{
  __shared__ unsigned short P[8][16][32];
  const int tid  = threadIdx.x, wave = tid >> 5, lane = tid & 31;
  const int half = lane >> 4, l16 = lane & 15;
  const int i0 = blockIdx.x * 128 + wave * 16;
  const int hd = blockIdx.y, b = blockIdx.z;
  const float scale = 0.17677669529663687f;  // 1/sqrt(32)

  FragB qf;
  {
    const unsigned short* qp = q + ((long)(b * 2048 + i0 + l16) * 256 + hd * 32);
#pragma unroll
    for (int v2 = 0; v2 < 4; ++v2) {
      qf.u[v2]     = *(const unsigned int*)(qp + 2 * v2 + half * 16);
      qf.u[4 + v2] = *(const unsigned int*)(qp + 8 + 2 * v2 + half * 16);
    }
  }
  float mrow[8], lrow[8];
  FragAcc o0, o1;
#pragma unroll
  for (int r = 0; r < 8; ++r) { mrow[r] = -1e30f; lrow[r] = 0.f; o0.f[r] = 0.f; o1.f[r] = 0.f; }

  const float* adjb = adj + (long)b * 2048 * 2048;

  for (int j0 = 0; j0 < 2048; j0 += 32) {
    // k fragments: B[d][j] = k[j][d]; dword packs (d, d+1) -> contiguous loads
    FragB kf0, kf1;
    {
      const unsigned short* kp0 = k + ((long)(b * 2048 + j0 + l16) * 256 + hd * 32 + half * 16);
      const unsigned short* kp1 = kp0 + 16 * 256;
#pragma unroll
      for (int v2 = 0; v2 < 8; ++v2) {
        kf0.u[v2] = *(const unsigned int*)(kp0 + 2 * v2);
        kf1.u[v2] = *(const unsigned int*)(kp1 + 2 * v2);
      }
    }
    FragAcc s0, s1;
#pragma unroll
    for (int r = 0; r < 8; ++r) { s0.f[r] = 0.f; s1.f[r] = 0.f; }
    s0.v = wmma_bf16(qf.v, kf0.v, s0.v);
    s1.v = wmma_bf16(qf.v, kf1.v, s1.v);

    // mask, scale, online softmax (per-row stats via 16-lane shuffles)
#pragma unroll
    for (int r = 0; r < 8; ++r) {
      int mg = i0 + r + half * 8;
      const float* arow = adjb + (long)mg * 2048 + j0 + l16;
      float sv0 = (arow[0]  > 0.f) ? s0.f[r] * scale : -1e30f;
      float sv1 = (arow[16] > 0.f) ? s1.f[r] * scale : -1e30f;
      float mcur = fmaxf(sv0, sv1);
      for (int off = 8; off >= 1; off >>= 1) mcur = fmaxf(mcur, __shfl_xor(mcur, off, 32));
      float mn = fmaxf(mrow[r], mcur);
      float alpha = __expf(mrow[r] - mn);
      float p0 = (sv0 > -1e29f) ? __expf(sv0 - mn) : 0.f;
      float p1 = (sv1 > -1e29f) ? __expf(sv1 - mn) : 0.f;
      float rs = p0 + p1;
      for (int off = 8; off >= 1; off >>= 1) rs += __shfl_xor(rs, off, 32);
      lrow[r] = lrow[r] * alpha + rs;
      mrow[r] = mn;
      o0.f[r] *= alpha; o1.f[r] *= alpha;
      P[wave][r + half * 8][l16]      = f2bf(p0);
      P[wave][r + half * 8][16 + l16] = f2bf(p1);
    }
    __builtin_amdgcn_wave_barrier();
    asm volatile("s_wait_dscnt 0" ::: "memory");

    // re-read p as A fragment
    FragB pf;
    {
      const unsigned short* pp = &P[wave][l16][0];
#pragma unroll
      for (int v2 = 0; v2 < 4; ++v2) {
        pf.u[v2]     = *(const unsigned int*)(pp + 2 * v2 + half * 16);
        pf.u[4 + v2] = *(const unsigned int*)(pp + 8 + 2 * v2 + half * 16);
      }
    }
    __builtin_amdgcn_wave_barrier();

    // v fragments from vT[b, c, n]: B[j][d]; dword packs (j, j+1) contiguous
    FragB vf0, vf1;
    {
      const unsigned short* vp0 = vT + ((long)b * 256 + hd * 32 + l16) * 2048 + j0 + half * 16;
      const unsigned short* vp1 = vp0 + 16 * 2048;
#pragma unroll
      for (int v2 = 0; v2 < 8; ++v2) {
        vf0.u[v2] = *(const unsigned int*)(vp0 + 2 * v2);
        vf1.u[v2] = *(const unsigned int*)(vp1 + 2 * v2);
      }
    }
    o0.v = wmma_bf16(pf.v, vf0.v, o0.v);
    o1.v = wmma_bf16(pf.v, vf1.v, o1.v);
  }

#pragma unroll
  for (int r = 0; r < 8; ++r) {
    float inv = 1.f / fmaxf(lrow[r], 1e-20f);
    int mg = i0 + r + half * 8;
    unsigned short* op = o + ((long)(b * 2048 + mg) * 256 + hd * 32);
    op[l16]      = f2bf(o0.f[r] * inv);
    op[16 + l16] = f2bf(o1.f[r] * inv);
  }
}

// ---------------------------------------------------------------------------
// Elementwise / normalization kernels
// ---------------------------------------------------------------------------
__global__ void cast_kernel(const float* __restrict__ x, unsigned short* __restrict__ y) {
  long i = (long)blockIdx.x * 256 + threadIdx.x;
  y[i] = f2bf(x[i]);
}

// GroupNorm stats: one block per (b, g); reduce over N*8 = 16384 elems.
__global__ __launch_bounds__(256) void gn_stats_kernel(const float* __restrict__ x,
                                                       float* __restrict__ stats) {
  __shared__ float ss[256], ss2[256];
  int bg = blockIdx.x;
  int b = bg >> 5, g = bg & 31;
  const float* xp = x + (long)b * 2048 * 256 + g * 8;
  float s = 0.f, s2 = 0.f;
  for (int idx = threadIdx.x; idx < 2048 * 8; idx += 256) {
    int n = idx >> 3, j = idx & 7;
    float v = xp[(long)n * 256 + j];
    s += v; s2 += v * v;
  }
  ss[threadIdx.x] = s; ss2[threadIdx.x] = s2;
  __syncthreads();
  for (int st = 128; st > 0; st >>= 1) {
    if (threadIdx.x < st) { ss[threadIdx.x] += ss[threadIdx.x + st]; ss2[threadIdx.x] += ss2[threadIdx.x + st]; }
    __syncthreads();
  }
  if (threadIdx.x == 0) {
    float m = ss[0] * (1.f / 16384.f);
    stats[bg * 2]     = m;
    stats[bg * 2 + 1] = ss2[0] * (1.f / 16384.f) - m * m;
  }
}

__global__ void gn_apply_kernel(const float* __restrict__ x, const float* __restrict__ stats,
                                const float* __restrict__ w, const float* __restrict__ bb,
                                float eps, int do_silu, unsigned short* __restrict__ outb) {
  long i = (long)blockIdx.x * 256 + threadIdx.x;
  int c = (int)(i & 255);
  int b = (int)(i >> 19);
  int g = c >> 3;
  float m = stats[(b * 32 + g) * 2], v = stats[(b * 32 + g) * 2 + 1];
  float y = (x[i] - m) * rsqrtf(v + eps) * w[c] + bb[c];
  if (do_silu) y = y / (1.f + __expf(-y));
  outb[i] = f2bf(y);
}

__global__ __launch_bounds__(256) void ln_kernel(const float* __restrict__ x,
                                                 const float* __restrict__ w,
                                                 const float* __restrict__ bb,
                                                 unsigned short* __restrict__ outb) {
  __shared__ float ss[256], ss2[256];
  __shared__ float mstat[2];
  long row = blockIdx.x;
  int t = threadIdx.x;
  float v = x[row * 256 + t];
  ss[t] = v; ss2[t] = v * v;
  __syncthreads();
  for (int st = 128; st > 0; st >>= 1) {
    if (t < st) { ss[t] += ss[t + st]; ss2[t] += ss2[t + st]; }
    __syncthreads();
  }
  if (t == 0) {
    float m = ss[0] * (1.f / 256.f);
    mstat[0] = m;
    mstat[1] = rsqrtf(ss2[0] * (1.f / 256.f) - m * m + 1e-5f);
  }
  __syncthreads();
  outb[row * 256 + t] = f2bf((v - mstat[0]) * mstat[1] * w[t] + bb[t]);
}

// t_emb projection: embvec[b,c] = silu(t_emb[b,:] @ emb_w[:,c] + emb_b[c])
__global__ void emb_kernel(const float* __restrict__ temb, const float* __restrict__ ew,
                           const float* __restrict__ eb, float* __restrict__ out) {
  int b = blockIdx.x, c = threadIdx.x;
  const float* t = temb + b * 1024;
  float acc = eb[c];
  for (int kk = 0; kk < 1024; ++kk) acc += t[kk] * ew[kk * 256 + c];
  out[b * 256 + c] = acc / (1.f + __expf(-acc));
}

// GEGLU: u = a * gelu_exact(gate), g = [.., 2048] split at 1024
__global__ void geglu_kernel(const float* __restrict__ g, unsigned short* __restrict__ u) {
  long i = (long)blockIdx.x * 256 + threadIdx.x;
  long row = i >> 10; int j = (int)(i & 1023);
  float a  = g[row * 2048 + j];
  float gt = g[row * 2048 + 1024 + j];
  float ge = 0.5f * gt * (1.f + erff(gt * 0.70710678118654752f));
  u[i] = f2bf(a * ge);
}

// ---------------------------------------------------------------------------
// Host orchestration
// ---------------------------------------------------------------------------
extern "C" void kernel_launch(void* const* d_in, const int* in_sizes, int n_in,
                              void* d_out, int out_size, void* d_ws, size_t ws_size,
                              hipStream_t stream) {
  (void)in_sizes; (void)n_in; (void)out_size; (void)ws_size;
  auto F = [&](int i) { return (const float*)d_in[i]; };

  const float* feat = F(0);
  const float* adjf = F(1);
  const float* temb = F(2);

  // --- workspace carve-up (~100 MB total)
  char* wp = (char*)d_ws;
  auto take = [&](long bytes) { void* r = wp; wp += (bytes + 255) & ~255L; return r; };
  unsigned short* adjb   = (unsigned short*)take(2L * 2048 * 2048 * 2);
  unsigned short* wb     = (unsigned short*)take(1703936L * 2);
  float*          stats  = (float*)take(64 * 2 * 4);
  float*          embvec = (float*)take(2 * 256 * 4);
  unsigned short* t0b    = (unsigned short*)take(2L * 2048 * 256 * 2);
  unsigned short* t1b    = (unsigned short*)take(2L * 2048 * 256 * 2);
  unsigned short* sb     = (unsigned short*)take(2L * 2048 * 256 * 2);
  unsigned short* qbuf   = (unsigned short*)take(2L * 2048 * 256 * 2);
  unsigned short* kbuf   = (unsigned short*)take(2L * 2048 * 256 * 2);
  unsigned short* vtb    = (unsigned short*)take(2L * 2048 * 256 * 2);
  unsigned short* obuf   = (unsigned short*)take(2L * 2048 * 256 * 2);
  unsigned short* hb16   = (unsigned short*)take(2L * 2048 * 256 * 2);
  float*          bufA   = (float*)take(2L * 2048 * 256 * 4);
  float*          bufH   = (float*)take(2L * 2048 * 256 * 4);
  float*          bufT   = (float*)take(2L * 2048 * 256 * 4);
  float*          gbuf   = (float*)take(2L * 2048 * 2048 * 4);
  unsigned short* ubuf   = (unsigned short*)take(2L * 2048 * 1024 * 2);
  unsigned short* subuf  = (unsigned short*)take(2L * 2048 * 1024 * 2);

  long wo = 0;
  auto wtake = [&](long n) { unsigned short* r = wb + wo; wo += n; return r; };
  unsigned short* w_r1_in  = wtake(65536);
  unsigned short* w_r1_out = wtake(65536);
  unsigned short* w_pi     = wtake(65536);
  unsigned short* w_a1q = wtake(65536); unsigned short* w_a1k = wtake(65536);
  unsigned short* w_a1v = wtake(65536); unsigned short* w_a1o = wtake(65536);
  unsigned short* w_a2q = wtake(65536); unsigned short* w_a2k = wtake(65536);
  unsigned short* w_a2v = wtake(65536); unsigned short* w_a2o = wtake(65536);
  unsigned short* w_geglu  = wtake(524288);
  unsigned short* w_ffo    = wtake(262144);
  unsigned short* w_po     = wtake(65536);
  unsigned short* w_r2_in  = wtake(65536);
  unsigned short* w_r2_out = wtake(65536);

  auto cast = [&](const float* src, unsigned short* dst, long n) {
    cast_kernel<<<(unsigned)(n / 256), 256, 0, stream>>>(src, dst);
  };

  // one-time casts: adjacency + all weight matrices -> bf16
  cast(adjf, adjb, 2L * 2048 * 2048);
  cast(F(5),  w_r1_in, 65536);  cast(F(11), w_r1_out, 65536);
  cast(F(15), w_pi, 65536);
  cast(F(19), w_a1q, 65536); cast(F(21), w_a1k, 65536);
  cast(F(23), w_a1v, 65536); cast(F(25), w_a1o, 65536);
  cast(F(29), w_a2q, 65536); cast(F(31), w_a2k, 65536);
  cast(F(33), w_a2v, 65536); cast(F(35), w_a2o, 65536);
  cast(F(39), w_geglu, 524288); cast(F(41), w_ffo, 262144);
  cast(F(43), w_po, 65536);
  cast(F(47), w_r2_in, 65536); cast(F(53), w_r2_out, 65536);

  auto gemm = [&](const void* A, long sAb, int lda, const void* B, long sBb, int ldb,
                  const float* bias, const float* resid, long sRb, const float* bvec,
                  float* Cf, long sCf, unsigned short* Cb, long sCb,
                  int Mr, int Nn, int K, int batches, int flags) {
    dim3 g(Nn / 128, Mr / 128, batches);
#define GEMM_CALL(FL) gemm_bf16_kernel<FL><<<g, 256, 0, stream>>>( \
        (const unsigned short*)A, sAb, lda, (const unsigned short*)B, sBb, ldb, \
        bias, resid, sRb, bvec, Cf, sCf, Cb, sCb, Nn, K)
    switch (flags) {
      case FLAG_WBF:                            GEMM_CALL(FLAG_WBF); break;
      case FLAG_BIAS | FLAG_WBF:                GEMM_CALL(FLAG_BIAS | FLAG_WBF); break;
      case FLAG_BIAS | FLAG_WBFT:               GEMM_CALL(FLAG_BIAS | FLAG_WBFT); break;
      case FLAG_BIAS | FLAG_BVEC | FLAG_WF32:   GEMM_CALL(FLAG_BIAS | FLAG_BVEC | FLAG_WF32); break;
      case FLAG_BIAS | FLAG_WF32:               GEMM_CALL(FLAG_BIAS | FLAG_WF32); break;
      case FLAG_BIAS | FLAG_RESID | FLAG_WF32:  GEMM_CALL(FLAG_BIAS | FLAG_RESID | FLAG_WF32); break;
      default: break;
    }
#undef GEMM_CALL
  };
  // adj @ X  (batched over B=2), bf16 output only
  auto spmm = [&](const void* actb, int nc, unsigned short* outb) {
    gemm(adjb, 2048L * 2048, 2048, actb, 2048L * nc, nc, nullptr, nullptr, 0, nullptr,
         nullptr, 0, outb, 2048L * nc, 2048, nc, 2048, 2, FLAG_WBF);
  };
  // dense [4096,K] @ [K,Nn]
  auto dense = [&](const void* inb, int K, const unsigned short* W, int Nn,
                   const float* bias, const float* resid, const float* bvec,
                   float* Cf, unsigned short* Cb, int flags) {
    (void)Cb;
    unsigned short* cb = (flags & (FLAG_WBF | FLAG_WBFT)) ? (unsigned short*)Cb : nullptr;
    gemm(inb, 0, K, W, 0, Nn, bias, resid, 0, bvec, Cf, 0, cb, 0, 4096, Nn, K, 1, flags);
  };

  auto resblock = [&](const float* x, float* hbuf, float* out,
                      const float* nw1, const float* nb1, const unsigned short* w1, const float* b1,
                      const float* embw, const float* embb,
                      const float* nw2, const float* nb2, const unsigned short* w2, const float* b2) {
    gn_stats_kernel<<<64, 256, 0, stream>>>(x, stats);
    gn_apply_kernel<<<4096, 256, 0, stream>>>(x, stats, nw1, nb1, 1e-5f, 1, t0b);
    spmm(t0b, 256, t1b);
    emb_kernel<<<2, 256, 0, stream>>>(temb, embw, embb, embvec);
    dense(t1b, 256, w1, 256, b1, nullptr, embvec, hbuf, nullptr,
          FLAG_BIAS | FLAG_BVEC | FLAG_WF32);
    gn_stats_kernel<<<64, 256, 0, stream>>>(hbuf, stats);
    gn_apply_kernel<<<4096, 256, 0, stream>>>(hbuf, stats, nw2, nb2, 1e-5f, 1, t0b);
    spmm(t0b, 256, t1b);
    dense(t1b, 256, w2, 256, b2, nullptr, nullptr, out, nullptr, FLAG_BIAS | FLAG_WF32);
  };

  auto mha = [&](float* h, const float* lnw, const float* lnbb,
                 const unsigned short* qw, const float* qbb,
                 const unsigned short* kw, const float* kbb,
                 const unsigned short* vw, const float* vbb,
                 const unsigned short* ow, const float* obb) {
    ln_kernel<<<4096, 256, 0, stream>>>(h, lnw, lnbb, t0b);
    spmm(t0b, 256, sb);                       // shared adj @ ln(x) for q,k,v
    dense(sb, 256, qw, 256, qbb, nullptr, nullptr, nullptr, qbuf, FLAG_BIAS | FLAG_WBF);
    dense(sb, 256, kw, 256, kbb, nullptr, nullptr, nullptr, kbuf, FLAG_BIAS | FLAG_WBF);
    dense(sb, 256, vw, 256, vbb, nullptr, nullptr, nullptr, vtb,  FLAG_BIAS | FLAG_WBFT);
    attn_kernel<<<dim3(16, 8, 2), 256, 0, stream>>>(qbuf, kbuf, vtb, adjf, obuf);
    spmm(obuf, 256, t1b);
    dense(t1b, 256, ow, 256, obb, h, nullptr, h, nullptr,
          FLAG_BIAS | FLAG_RESID | FLAG_WF32);
  };

  // ---------------- res1: feat -> bufA ----------------
  resblock(feat, bufH, bufA, F(3), F(4), w_r1_in, F(6), F(7), F(8),
           F(9), F(10), w_r1_out, F(12));

  // ---------------- transformer: bufA -> bufT ----------------
  gn_stats_kernel<<<64, 256, 0, stream>>>(bufA, stats);
  gn_apply_kernel<<<4096, 256, 0, stream>>>(bufA, stats, F(13), F(14), 1e-6f, 0, t0b);
  spmm(t0b, 256, t1b);
  dense(t1b, 256, w_pi, 256, F(16), nullptr, nullptr, bufH, nullptr, FLAG_BIAS | FLAG_WF32);

  mha(bufH, F(17), F(18), w_a1q, F(20), w_a1k, F(22), w_a1v, F(24), w_a1o, F(26));
  mha(bufH, F(27), F(28), w_a2q, F(30), w_a2k, F(32), w_a2v, F(34), w_a2o, F(36));

  ln_kernel<<<4096, 256, 0, stream>>>(bufH, F(37), F(38), t0b);
  spmm(t0b, 256, sb);
  dense(sb, 256, w_geglu, 2048, F(40), nullptr, nullptr, gbuf, nullptr, FLAG_BIAS | FLAG_WF32);
  geglu_kernel<<<16384, 256, 0, stream>>>(gbuf, ubuf);
  spmm(ubuf, 1024, subuf);
  dense(subuf, 1024, w_ffo, 256, F(42), bufH, nullptr, bufH, nullptr,
        FLAG_BIAS | FLAG_RESID | FLAG_WF32);

  cast(bufH, hb16, 2L * 2048 * 256);
  spmm(hb16, 256, t1b);
  dense(t1b, 256, w_po, 256, F(44), bufA, nullptr, bufT, nullptr,
        FLAG_BIAS | FLAG_RESID | FLAG_WF32);

  // ---------------- res2: bufT -> d_out ----------------
  resblock(bufT, bufH, (float*)d_out, F(45), F(46), w_r2_in, F(48), F(49), F(50),
           F(51), F(52), w_r2_out, F(54));
}
